// DAGH_sample_loss_45440753991782
// MI455X (gfx1250) — matrix operand order repaired and checked by reading, not compiled, not run on hardware
//
#include <hip/hip_runtime.h>

#define CODE_K 64

typedef __attribute__((ext_vector_type(16))) __bf16 v16bf;
typedef __attribute__((ext_vector_type(8)))  __bf16 v8bf;
typedef __attribute__((ext_vector_type(8)))  float  v8f;

__device__ __forceinline__ v16bf join16(v8bf lo, v8bf hi) {
    v16bf r;
#pragma unroll
    for (int t = 0; t < 8; ++t) { r[t] = lo[t]; r[t + 8] = hi[t]; }
    return r;
}

// ---------------------------------------------------------------------------
// Kernel 1: transpose F,B -> bf16 [n,64]; column norms fn,bn; zero accumulator
// ---------------------------------------------------------------------------
__global__ void __launch_bounds__(256)
prep_kernel(const float* __restrict__ F, const float* __restrict__ B,
            float* __restrict__ fn, float* __restrict__ bn,
            __bf16* __restrict__ Ft, __bf16* __restrict__ Bt,
            float* __restrict__ tr_acc, int n)
{
    int i = blockIdx.x * blockDim.x + threadIdx.x;
    if (i == 0) *tr_acc = 0.0f;
    if (i >= n) return;
    float fa = 0.0f, ba = 0.0f;
    __bf16 fr[CODE_K], br[CODE_K];
#pragma unroll
    for (int k = 0; k < CODE_K; ++k) {          // coalesced: consecutive i per row k
        float f = F[(size_t)k * n + i];
        float b = B[(size_t)k * n + i];
        fa += f * f;  ba += b * b;
        fr[k] = (__bf16)f;  br[k] = (__bf16)b;
    }
    fn[i] = fa;  bn[i] = ba;
    v8bf* fd = (v8bf*)(Ft + (size_t)i * CODE_K);
    v8bf* bd = (v8bf*)(Bt + (size_t)i * CODE_K);
#pragma unroll
    for (int c = 0; c < 8; ++c) {
        v8bf vf, vb;
#pragma unroll
        for (int t = 0; t < 8; ++t) { vf[t] = fr[c * 8 + t]; vb[t] = br[c * 8 + t]; }
        fd[c] = vf;  bd[c] = vb;
    }
}

// ---------------------------------------------------------------------------
// Kernel 2: single pass over w.  tr += w_ij * (fn_i + bn_j - 2*(F_i . B_j))
// D tiles via v_wmma_f32_16x16x32_bf16; two INDEPENDENT K-chunk accumulators
// (no WMMA->WMMA RAW, no hazard nops), summed in the VALU fold.
// ---------------------------------------------------------------------------
__global__ void __launch_bounds__(256)
tr_kernel(const float* __restrict__ w,
          const __bf16* __restrict__ Ft, const __bf16* __restrict__ Bt,
          const float* __restrict__ fn, const float* __restrict__ bn,
          float* __restrict__ tr_acc, int n)
{
    const int lane   = threadIdx.x & 31;
    const int wave   = threadIdx.x >> 5;        // 0..7
    const int laneN  = lane & 15;               // N / column within tile
    const int laneHi = lane >> 4;               // 0 | 1
    const int i0     = blockIdx.x * 128 + wave * 16;
    const int jcount = n / (16 * gridDim.y);
    const int jbase  = blockIdx.y * jcount * 16;

    // A fragment (F^T tile rows i0..i0+15, K=0..63), loaded once per wave.
    // 16-bit A layout: lane M = laneN; lo-lanes hold K {0..7,16..23}(+32c), hi-lanes +8.
    const __bf16* frow = Ft + (size_t)(i0 + laneN) * CODE_K;
    const int aoff = laneHi * 8;
    v16bf a0 = join16(*(const v8bf*)(frow + aoff),      *(const v8bf*)(frow + aoff + 16));
    v16bf a1 = join16(*(const v8bf*)(frow + 32 + aoff), *(const v8bf*)(frow + 32 + aoff + 16));

    float fnv[8];
#pragma unroll
    for (int r = 0; r < 8; ++r) fnv[r] = fn[i0 + laneHi * 8 + r];

    // strength-reduced bases for the j loop
    const float*  wr0 = w  + (size_t)(i0 + laneHi * 8) * n + jbase + laneN;
    const __bf16* br0 = Bt + (size_t)(jbase + laneN) * CODE_K + laneHi * 8;
    const float*  bnp = bn + jbase + laneN;

    float acc0 = 0.0f, acc1 = 0.0f, acc2 = 0.0f, acc3 = 0.0f;
    for (int t = 0; t < jcount; ++t) {
        const float bv = bnp[t * 16];
        // B fragment: lane N = laneN; lo-lanes K {0..7,16..23} (+32c), hi-lanes +8
        const __bf16* br = br0 + (size_t)t * 16 * CODE_K;
        v16bf b0 = join16(*(const v8bf*)(br),      *(const v8bf*)(br + 16));
        v16bf b1 = join16(*(const v8bf*)(br + 32), *(const v8bf*)(br + 48));

        const v8f z = {};
        v8f d0 = __builtin_amdgcn_wmma_f32_16x16x32_bf16(false, a0, false, b0, (short)0, z, false, false);
        v8f d1 = __builtin_amdgcn_wmma_f32_16x16x32_bf16(false, a1, false, b1, (short)0, z, false, false);

        // C/D layout: VGPR r, lane l -> M = 8*(l>=16)+r, N = l&15
        const float* wr = wr0 + (size_t)t * 16;
#pragma unroll
        for (int r = 0; r < 8; ++r) {
            float wv   = __builtin_nontemporal_load(&wr[(size_t)r * n]);  // single-use stream
            float term = wv * (fnv[r] + bv - 2.0f * (d0[r] + d1[r]));
            if      ((r & 3) == 0) acc0 += term;
            else if ((r & 3) == 1) acc1 += term;
            else if ((r & 3) == 2) acc2 += term;
            else                   acc3 += term;
        }
    }
    float acc = (acc0 + acc1) + (acc2 + acc3);

    __shared__ float red[256];
    red[threadIdx.x] = acc;
    __syncthreads();
    for (int s = 128; s > 0; s >>= 1) {
        if (threadIdx.x < s) red[threadIdx.x] += red[threadIdx.x + s];
        __syncthreads();
    }
    if (threadIdx.x == 0) atomicAdd(tr_acc, red[0]);
}

// ---------------------------------------------------------------------------
// Kernel 3 (1 block): gram = F F^T via WMMA -> oth; balance term; final loss
// ---------------------------------------------------------------------------
__global__ void __launch_bounds__(256)
finalize_kernel(const float* __restrict__ F, const float* __restrict__ tr_acc,
                float* __restrict__ out, int n)
{
    __shared__ float sred[256];
    const int tid    = threadIdx.x;
    const int lane   = tid & 31;
    const int wave   = tid >> 5;
    const int laneN  = lane & 15;
    const int laneHi = lane >> 4;

    // ---- orthogonality: 16 gram tiles (64x64), 2 per wave, K = n via bf16 WMMA
    float othp = 0.0f;
    const float inv_n = 1.0f / (float)n;
#pragma unroll
    for (int tt = 0; tt < 2; ++tt) {
        const int tile = wave * 2 + tt;
        const int a0 = (tile >> 2) * 16;
        const int b0 = (tile & 3) * 16;
        v8f c = {};
        for (int i0 = 0; i0 < n; i0 += 32) {
            const float* ap = F + (size_t)(a0 + laneN) * n + i0 + laneHi * 8;
            v16bf av;
#pragma unroll
            for (int q = 0; q < 8; ++q) { av[q] = (__bf16)ap[q]; av[q + 8] = (__bf16)ap[q + 16]; }
            const float* bp = F + (size_t)(b0 + laneN) * n + i0 + laneHi * 8;
            v16bf bv;
#pragma unroll
            for (int q = 0; q < 8; ++q) { bv[q] = (__bf16)bp[q]; bv[q + 8] = (__bf16)bp[q + 16]; }
            c = __builtin_amdgcn_wmma_f32_16x16x32_bf16(false, av, false, bv, (short)0, c, false, false);
        }
#pragma unroll
        for (int r = 0; r < 8; ++r) {
            int ga = a0 + laneHi * 8 + r;
            int gb = b0 + laneN;
            float v = c[r] * inv_n - (ga == gb ? 1.0f : 0.0f);
            othp += v * v;
        }
    }
    sred[tid] = othp;
    __syncthreads();
    for (int s = 128; s > 0; s >>= 1) { if (tid < s) sred[tid] += sred[tid + s]; __syncthreads(); }
    const float oth = sred[0];
    __syncthreads();

    // ---- balance: rs[k] = sum_i F[k,i]; bla = sum_k rs^2
    const int k   = tid & 63;
    const int seg = tid >> 6;                   // 4 segments per row
    const int seglen = n >> 2;
    const float* fr = F + (size_t)k * n + seg * seglen;
    float p = 0.0f;
    for (int ii = 0; ii < seglen; ++ii) p += fr[ii];
    sred[tid] = p;
    __syncthreads();
    if (tid < 64) {
        float rs = sred[tid] + sred[tid + 64] + sred[tid + 128] + sred[tid + 192];
        sred[tid] = rs * rs;
    }
    __syncthreads();
    for (int s = 32; s > 0; s >>= 1) { if (tid < s) sred[tid] += sred[tid + s]; __syncthreads(); }

    if (tid == 0) {
        const float bla = sred[0];
        const float tr  = *tr_acc;
        const float nf  = (float)n;
        out[0] = 0.5f * tr / (nf * nf) * 10000.0f
               + 0.5f * bla / nf
               + 0.5f * oth / (float)CODE_K;
    }
}

// ---------------------------------------------------------------------------
extern "C" void kernel_launch(void* const* d_in, const int* in_sizes, int n_in,
                              void* d_out, int out_size, void* d_ws, size_t ws_size,
                              hipStream_t stream)
{
    const float* w = (const float*)d_in[0];
    const float* F = (const float*)d_in[1];
    const float* B = (const float*)d_in[2];
    const int n = in_sizes[1] / CODE_K;         // 8192

    // workspace layout: [tr_acc pad16][fn n][bn n][Ft bf16 n*64][Bt bf16 n*64]
    float*  tr_acc = (float*)d_ws;
    float*  fn     = (float*)d_ws + 16;
    float*  bn     = fn + n;
    __bf16* Ft     = (__bf16*)(bn + n);
    __bf16* Bt     = Ft + (size_t)n * CODE_K;

    prep_kernel<<<(n + 255) / 256, 256, 0, stream>>>(F, B, fn, bn, Ft, Bt, tr_acc, n);

    dim3 grid(n / 128, 8);                      // 512 blocks, 4096 waves
    tr_kernel<<<grid, 256, 0, stream>>>(w, Ft, Bt, fn, bn, tr_acc, n);

    finalize_kernel<<<1, 256, 0, stream>>>(F, tr_acc, (float*)d_out, n);
}
